// RecurrentDecoder_9320079033275
// MI455X (gfx1250) — matrix operand order repaired and checked by reading, not compiled, not run on hardware
//
#include <hip/hip_runtime.h>
#include <hip/hip_bf16.h>
#include <math.h>

// Problem constants (B,N,D,Z,H) = (256,128,256,64,128), pred_len=30
#define BNROWS 32768   // B*N
#define DDIM   256
#define ZDIM   64
#define HDIM   128
#define G3     384     // 3*H
#define KIN    320     // D+Z

typedef __attribute__((ext_vector_type(16))) __bf16 v16bf;
typedef __attribute__((ext_vector_type(8)))  float  v8f;

// ---------------------------------------------------------------------------
// WMMA v_wmma_f32_16x16x32_bf16 fragment layouts as used here (wave32, contiguous k, A and B match):
//  A (16x32 MxK): lanes 0-15 -> M=lane, elems 0..15 = K kb..kb+15,
//                 kb = kc*32;
//                 lanes 16-31 -> same M, kb += 16.
//  B (32x16 KxN): lanes 0-15 -> N=lane, K kc*32+0..15 ; lanes 16-31 -> K +16..31.
//  C/D (16x16 f32): lane l, elem r -> (M = r + 8*(l>>4), N = l&15).
//
// A and B tiles are staged in LDS pre-swizzled into these layouts so each
// fragment is one 32-byte contiguous LDS vector load per lane.
// ---------------------------------------------------------------------------

// swizzled element slot for A-fragment storage: buffer [mt][kc][lane][e]
__device__ __forceinline__ int a_slot(int m, int k, int KC) {
  int mt = m >> 4, kc = k >> 5, kk = k & 31;
  int lane = (m & 15) + ((kk & 16) ? 16 : 0);
  int e    = kk & 15;
  return ((mt * KC + kc) * 32 + lane) * 16 + e;
}

// swizzled element slot for B-fragment storage: buffer [nt][kc][lane][e]
__device__ __forceinline__ int b_slot(int k, int n, int KC) {
  int nt = n >> 4, kc = k >> 5;
  int lane = (n & 15) + ((k & 16) ? 16 : 0);
  int e    = k & 15;
  return ((nt * KC + kc) * 32 + lane) * 16 + e;
}

// B fragment gathered straight from global, W row-major (N x K): B[k][n]=W[n*K+k]
__device__ __forceinline__ v16bf load_b_from_wT(const float* __restrict__ W,
                                                int K, int ntile, int kc,
                                                int lane) {
  int n  = ntile * 16 + (lane & 15);
  int kb = kc * 32 + ((lane & 16) ? 16 : 0);
  const float* p = W + (size_t)n * K + kb;   // 16 contiguous f32 (64B/lane)
  v16bf b;
#pragma unroll
  for (int e = 0; e < 16; ++e) b[e] = (__bf16)p[e];
  return b;
}

// B fragment from global, W row-major (K x N): B[k][n]=W[k*N+n]
__device__ __forceinline__ v16bf load_b_from_w(const float* __restrict__ W,
                                               int N, int ntile, int kc,
                                               int lane) {
  int n  = ntile * 16 + (lane & 15);
  int kb = kc * 32 + ((lane & 16) ? 16 : 0);
  const float* p = W + (size_t)kb * N + n;
  v16bf b;
#pragma unroll
  for (int e = 0; e < 16; ++e) b[e] = (__bf16)p[(size_t)e * N];
  return b;
}

// Fast gate nonlinearities: v_exp_f32 + v_rcp_f32
__device__ __forceinline__ float fast_rcp(float x) {
  return __builtin_amdgcn_rcpf(x);
}
__device__ __forceinline__ float sigm(float x) {
  return fast_rcp(1.0f + __expf(-x));
}
__device__ __forceinline__ float tanh_fast(float x) {
  float e = __expf(-2.0f * x);
  return (1.0f - e) * fast_rcp(1.0f + e);
}

// CDNA5 async global->LDS 16B copy (tracked by ASYNCcnt, no VGPR round-trip).
// LDS byte address = low 32 bits of the generic pointer (LDS aperture is in
// addr[63:32] on gfx1250, so truncation yields the LDS offset).
__device__ __forceinline__ void async_copy_b128(void* lds_dst,
                                                const void* gsrc) {
  unsigned ldsa = (unsigned)(uintptr_t)lds_dst;
  unsigned long long ga = (unsigned long long)(uintptr_t)gsrc;
  asm volatile("global_load_async_to_lds_b128 %0, %1, off"
               :: "v"(ldsa), "v"(ga)
               : "memory");
}
__device__ __forceinline__ void async_wait0() {
  asm volatile("s_wait_asynccnt 0" ::: "memory");
}

// ---------------------------------------------------------------------------
// Kernel 1: base_proj[BN,128] = concat(h_obs, z) @ W_in[0:320] + b_in
// Block = 64 rows; wave w owns N-tile w; K = 320 (10 chunks of 32)
// ---------------------------------------------------------------------------
__global__ __launch_bounds__(256) void k_base_proj(
    const float* __restrict__ h_obs, const float* __restrict__ z,
    const float* __restrict__ W_in, const float* __restrict__ b_in,
    float* __restrict__ base_proj) {
  __shared__ v16bf shAf[4 * 10 * 32];   // A fragments, 40 KB
  __shared__ float shBin[HDIM];

  const int tid  = threadIdx.x;
  const int lane = tid & 31;
  const int wave = tid >> 5;
  const int row0 = blockIdx.x * 64;

  __bf16* shAe = (__bf16*)shAf;
  for (int idx = tid; idx < 64 * KIN; idx += 256) {
    int m = idx / KIN, k = idx - m * KIN;
    int rn = row0 + m;
    float v = (k < DDIM) ? h_obs[(size_t)rn * DDIM + k]
                         : z[(size_t)rn * ZDIM + (k - DDIM)];
    shAe[a_slot(m, k, 10)] = (__bf16)v;
  }
  if (tid < HDIM) shBin[tid] = b_in[tid];
  __syncthreads();

  v8f acc[4] = {};
#pragma unroll 2
  for (int kc = 0; kc < 10; ++kc) {
    v16bf bf = load_b_from_w(W_in, HDIM, wave, kc, lane);
#pragma unroll
    for (int mt = 0; mt < 4; ++mt) {
      v16bf af = shAf[(mt * 10 + kc) * 32 + lane];
      acc[mt] = __builtin_amdgcn_wmma_f32_16x16x32_bf16(
          false, af, false, bf, (short)0, acc[mt], false, false);
    }
  }

  const int n  = wave * 16 + (lane & 15);
  const int mo = 8 * (lane >> 4);
  const float bb = shBin[n];
#pragma unroll
  for (int mt = 0; mt < 4; ++mt)
#pragma unroll
    for (int r = 0; r < 8; ++r) {
      int m = mt * 16 + r + mo;
      base_proj[(size_t)(row0 + m) * HDIM + n] = acc[mt][r] + bb;
    }
}

// ---------------------------------------------------------------------------
// Kernel 2: M2[2][384] = W_pos @ W_ih^T  (W_pos = rows 320..321 of W_in)
// ---------------------------------------------------------------------------
__global__ void k_m2(const float* __restrict__ W_in,
                     const float* __restrict__ W_ih,
                     float* __restrict__ M2w) {
  int idx = blockIdx.x * blockDim.x + threadIdx.x;
  if (idx >= 2 * G3) return;
  int c = idx & 1, j = idx >> 1;
  const float* wp = W_in + (size_t)(KIN + c) * HDIM;
  const float* wi = W_ih + (size_t)j * HDIM;
  float s = 0.f;
#pragma unroll 4
  for (int k = 0; k < HDIM; ++k) s += wp[k] * wi[k];
  M2w[c * G3 + j] = s;
}

// ---------------------------------------------------------------------------
// Kernel 3: G[BN,384] = base_proj @ W_ih^T + b_ih (+ b_hh for r/z gates only)
// b_hh is additive inside sigmoid for r/z, but for the n-gate b_hh is scaled
// by r (n = tanh(i_n + r*(h@W_hh^T + b_hh)_n)), so it must stay separate.
// Block = 64 rows; wave w owns N-tiles {w, w+8, w+16}; K = 128 (4 chunks)
// ---------------------------------------------------------------------------
__global__ __launch_bounds__(256) void k_gi(
    const float* __restrict__ base_proj, const float* __restrict__ W_ih,
    const float* __restrict__ b_ih, const float* __restrict__ b_hh,
    float* __restrict__ G) {
  __shared__ v16bf shAf[4 * 4 * 32];    // A fragments, 16 KB
  __shared__ float shBias[G3];

  const int tid  = threadIdx.x;
  const int lane = tid & 31;
  const int wave = tid >> 5;
  const int row0 = blockIdx.x * 64;

  __bf16* shAe = (__bf16*)shAf;
  for (int idx = tid; idx < 64 * HDIM; idx += 256) {
    int m = idx >> 7, k = idx & 127;
    shAe[a_slot(m, k, 4)] = (__bf16)base_proj[(size_t)(row0 + m) * HDIM + k];
  }
  for (int idx = tid; idx < G3; idx += 256)
    shBias[idx] = b_ih[idx] + ((idx < 2 * HDIM) ? b_hh[idx] : 0.f);
  __syncthreads();

  v8f acc[4][3] = {};
#pragma unroll
  for (int kc = 0; kc < 4; ++kc) {
    v16bf bf[3];
#pragma unroll
    for (int g = 0; g < 3; ++g)
      bf[g] = load_b_from_wT(W_ih, HDIM, wave + 8 * g, kc, lane);
#pragma unroll
    for (int mt = 0; mt < 4; ++mt) {
      v16bf af = shAf[(mt * 4 + kc) * 32 + lane];
#pragma unroll
      for (int g = 0; g < 3; ++g)
        acc[mt][g] = __builtin_amdgcn_wmma_f32_16x16x32_bf16(
            false, af, false, bf[g], (short)0, acc[mt][g], false, false);
    }
  }

  const int jl = wave * 16 + (lane & 15);
  const int mo = 8 * (lane >> 4);
#pragma unroll
  for (int g = 0; g < 3; ++g) {
    int n = jl + 128 * g;
    float bb = shBias[n];
#pragma unroll
    for (int mt = 0; mt < 4; ++mt)
#pragma unroll
      for (int r = 0; r < 8; ++r) {
        int m = mt * 16 + r + mo;
        G[(size_t)(row0 + m) * G3 + n] = acc[mt][g][r] + bb;
      }
  }
}

// ---------------------------------------------------------------------------
// Kernel 4: persistent recurrent GRU. Block = 64 rows for all pred_len steps.
// Everything the loop touches is LDS-resident:
//   - G tile staged via async global->LDS B128 (ASYNCcnt), overlapped with
//     the W_hh^T swizzle/convert staging
//   - W_hh^T staged once, pre-swizzled in B-fragment layout (96 KB)
//   - h kept as pre-swizzled bf16 A-fragments + f32 copy for the delta dot
// Per step: 28 LDS vector loads + 48 WMMAs per wave + register gate math.
// Only per-step global op: the trajectory store.
// ---------------------------------------------------------------------------
__global__ __launch_bounds__(256) void k_recurrent(
    const float* __restrict__ G, const float* __restrict__ M2w,
    const float* __restrict__ W_hh, const float* __restrict__ b_hh,
    const float* __restrict__ W_out, const float* __restrict__ b_out,
    const float* __restrict__ start_pos, const int* __restrict__ pred_len_p,
    float* __restrict__ out) {
  __shared__ v16bf shBf[24 * 4 * 32];      // W_hh^T B-fragments, 96 KB
  __shared__ float sh_G[64][G3 + 4];       // G tile (gi minus pos term), 97 KB
  __shared__ v16bf sh_hA[4 * 4 * 32];      // h as A-fragments (bf16), 16 KB
  __shared__ float sh_h[64][HDIM + 4];     // h f32 (delta phase), 33.8 KB
  __shared__ float sh_wout[2][HDIM];
  __shared__ float sh_pos[64][2];
  __shared__ float sh_bout[2];

  const int tid  = threadIdx.x;
  const int lane = tid & 31;
  const int wave = tid >> 5;
  const int row0 = blockIdx.x * 64;
  const int T    = *pred_len_p;

  // ---- One-time staging ----
  // G tile: async global->LDS 16B transfers; overlaps with the staging below.
  for (int idx = tid; idx < 64 * (G3 / 4); idx += 256) {
    int m = idx / (G3 / 4), q = idx - m * (G3 / 4);
    async_copy_b128(&sh_G[m][q * 4], G + (size_t)(row0 + m) * G3 + q * 4);
  }
  __bf16* shBe = (__bf16*)shBf;
  for (int idx = tid; idx < G3 * HDIM; idx += 256) {
    int n = idx >> 7, k = idx & 127;              // W_hh^T[k][n] = W_hh[n][k]
    shBe[b_slot(k, n, 4)] = (__bf16)W_hh[idx];    // coalesced global read
  }
  for (int idx = tid; idx < 4 * 4 * 32 * 8; idx += 256)
    ((int*)sh_hA)[idx] = 0;                       // h0 = 0 (bf16 frags)
  for (int idx = tid; idx < 64 * (HDIM + 4); idx += 256)
    (&sh_h[0][0])[idx] = 0.f;                     // h0 = 0 (f32)
  for (int idx = tid; idx < 2 * HDIM; idx += 256) sh_wout[0][idx] = W_out[idx];
  if (tid < 2) sh_bout[tid] = b_out[tid];
  if (tid < 128) {
    int m = tid >> 1, c = tid & 1;
    sh_pos[m][c] = start_pos[(size_t)(row0 + m) * 2 + c];
  }

  // Per-lane constants: column jl owns gates at jl + 128*g
  const int jl = wave * 16 + (lane & 15);
  const int mo = 8 * (lane >> 4);
  float m2x[3], m2y[3];
#pragma unroll
  for (int g = 0; g < 3; ++g) {
    m2x[g] = M2w[jl + 128 * g];
    m2y[g] = M2w[G3 + jl + 128 * g];
  }
  const float bhn = b_hh[jl + 2 * HDIM];   // n-gate hidden bias (scaled by r)

  // Swizzled write slot constants for this lane's column jl
  const int kcl     = jl >> 5;
  const int kk      = jl & 31;
  const int lane_hi = (kk & 16) ? 16 : 0;
  const int eA      = kk & 15;
  __bf16* hAe = (__bf16*)sh_hA;

  async_wait0();      // G tile landed in LDS
  __syncthreads();

  for (int t = 0; t < T; ++t) {
    // --- gh = h @ W_hh^T : 28 LDS vector loads + 48 WMMAs per wave ---
    v8f acc[4][3] = {};
#pragma unroll
    for (int kc = 0; kc < 4; ++kc) {
      v16bf bf[3];
#pragma unroll
      for (int g = 0; g < 3; ++g)
        bf[g] = shBf[((wave + 8 * g) * 4 + kc) * 32 + lane];
#pragma unroll
      for (int mt = 0; mt < 4; ++mt) {
        v16bf af = sh_hA[(mt * 4 + kc) * 32 + lane];
#pragma unroll
        for (int g = 0; g < 3; ++g)
          acc[mt][g] = __builtin_amdgcn_wmma_f32_16x16x32_bf16(
              false, af, false, bf[g], (short)0, acc[mt][g], false, false);
      }
    }
    __syncthreads();  // all reads of sh_hA complete

    // --- gates + state update in registers; write h (bf16 frag + f32) ---
#pragma unroll
    for (int mt = 0; mt < 4; ++mt) {
#pragma unroll
      for (int r = 0; r < 8; ++r) {
        int m = mt * 16 + r + mo;
        float2 pp = *(const float2*)&sh_pos[m][0];
        const float* Gr = &sh_G[m][0];
        float gir = Gr[jl]       + pp.x * m2x[0] + pp.y * m2y[0];
        float giz = Gr[jl + 128] + pp.x * m2x[1] + pp.y * m2y[1];
        float gin = Gr[jl + 256] + pp.x * m2x[2] + pp.y * m2y[2];
        float rg = sigm(gir + acc[mt][0][r]);
        float zg = sigm(giz + acc[mt][1][r]);
        float ng = tanh_fast(gin + rg * (acc[mt][2][r] + bhn));
        float ho = sh_h[m][jl];
        float hn = (1.0f - zg) * ng + zg * ho;
        sh_h[m][jl] = hn;
        hAe[((mt * 4 + kcl) * 32 + (m & 15) + lane_hi) * 16 + eA] = (__bf16)hn;
      }
    }
    __syncthreads();  // new h visible

    // --- delta = h @ W_out^T + b_out ; pos += delta ; emit trajectory ---
    if (tid < 128) {
      int m = tid >> 1, c = tid & 1;
      const float4* hv = (const float4*)&sh_h[m][0];      // 16B-aligned rows
      const float4* wv = (const float4*)&sh_wout[c][0];
      float d = sh_bout[c];
#pragma unroll
      for (int j = 0; j < HDIM / 4; ++j) {
        float4 a = hv[j], b = wv[j];
        d += a.x * b.x + a.y * b.y + a.z * b.z + a.w * b.w;
      }
      float p = sh_pos[m][c] + d;
      sh_pos[m][c] = p;
      int rn = row0 + m;
      int bb = rn >> 7;         // batch index (N=128)
      int nn = rn & 127;        // agent index
      out[(((size_t)bb * T + t) * 128 + nn) * 2 + c] = p;
    }
    __syncthreads();  // pos update visible for next step
  }
}

// ---------------------------------------------------------------------------
extern "C" void kernel_launch(void* const* d_in, const int* in_sizes, int n_in,
                              void* d_out, int out_size, void* d_ws,
                              size_t ws_size, hipStream_t stream) {
  const float* h_obs     = (const float*)d_in[0];
  const float* z         = (const float*)d_in[1];
  const float* start_pos = (const float*)d_in[2];
  const float* W_in      = (const float*)d_in[3];
  const float* b_in      = (const float*)d_in[4];
  const float* W_ih      = (const float*)d_in[5];
  const float* b_ih      = (const float*)d_in[6];
  const float* W_hh      = (const float*)d_in[7];
  const float* b_hh      = (const float*)d_in[8];
  const float* W_out     = (const float*)d_in[9];
  const float* b_out     = (const float*)d_in[10];
  const int*   pred_len  = (const int*)d_in[11];
  float* out = (float*)d_out;

  float* ws  = (float*)d_ws;
  float* Gm  = ws;                                   // BN*384 f32 (~50 MB)
  float* bp  = Gm + (size_t)BNROWS * G3;             // BN*128 f32 (~17 MB)
  float* M2w = bp + (size_t)BNROWS * HDIM;           // 768 f32

  k_base_proj<<<BNROWS / 64, 256, 0, stream>>>(h_obs, z, W_in, b_in, bp);
  k_m2<<<3, 256, 0, stream>>>(W_in, W_ih, M2w);
  k_gi<<<BNROWS / 64, 256, 0, stream>>>(bp, W_ih, b_ih, b_hh, Gm);
  k_recurrent<<<BNROWS / 64, 256, 0, stream>>>(Gm, M2w, W_hh, b_hh, W_out,
                                               b_out, start_pos, pred_len, out);
}
